// DotProductionAttention_56495999812159
// MI455X (gfx1250) — matrix operand order repaired and checked by reading, not compiled, not run on hardware
//
#include <hip/hip_runtime.h>
#include <stdint.h>

// ---------------------------------------------------------------------------
// Problem constants (reference: B=8, LQ=1024, LK=2048, D=1024, fp32)
// ---------------------------------------------------------------------------
#define BN   8
#define LQN  1024
#define LKN  2048
#define DIMN 1024
#define TD   64      // M/N output tile per workgroup (8 waves, 64x64 tile)
#define KC   64      // K chunk staged in LDS per iteration
#define LDP  72      // padded LDS row pitch (bf16): 144B rows -> conflict-free b128

typedef __attribute__((ext_vector_type(8)))  __bf16 v8bf;
typedef __attribute__((ext_vector_type(16))) __bf16 v16bf;
typedef __attribute__((ext_vector_type(8)))  float  v8f;

// round-to-nearest f32->bf16 pair packed in one dword via v_perm_b32:
//   round both mantissas (+0x8000), then one v_perm grabs both high halves.
__device__ __forceinline__ uint32_t pkbf(float f0, float f1) {
  const uint32_t u0 = __builtin_bit_cast(uint32_t, f0) + 0x8000u;
  const uint32_t u1 = __builtin_bit_cast(uint32_t, f1) + 0x8000u;
  return __builtin_amdgcn_perm(u1, u0, 0x07060302u);  // {u1[31:16], u0[31:16]}
}

__device__ __forceinline__ __bf16 f2bf(float f) {
  const uint32_t u = (__builtin_bit_cast(uint32_t, f) + 0x8000u) >> 16;
  union { uint16_t s; __bf16 b; } o; o.s = (uint16_t)u;
  return o.b;
}

#define CAT16(lo, hi) __builtin_shufflevector(lo, hi, 0,1,2,3,4,5,6,7,8,9,10,11,12,13,14,15)

// ---------------------------------------------------------------------------
// Core WMMA micro-tile: one wave computes 16 rows x 32 cols of C.
//   As  : [64 rows m][LDP]  (M-major, K contiguous)
//   BsT : [64 rows n][LDP]  (N-major, K contiguous)
// Fragment mapping (wave32):
//   A: lane m=lane&15, half=lane>>4: elem j -> K=half*16+j (contiguous in As row)
//   B: lane n=lane&15, half=lane>>4: elem j -> K=half*16+j (contiguous in BsT row)
//   C/D: VGPR r -> row (r + 8*half), col = lane&15
// All LDS reads are aligned 16B -> ds_load_b128, conflict-free with LDP=72.
// ---------------------------------------------------------------------------
__device__ __forceinline__ void wmma_tile(const __bf16* As, const __bf16* BsT,
                                          int mr, int nc0, v8f& acc0, v8f& acc1) {
  const int lane = threadIdx.x & 31;
  const int lm   = lane & 15;
  const int hf   = lane >> 4;
  const __bf16* Arow  = As  + (mr + lm) * LDP;
  const __bf16* B0row = BsT + (nc0 + lm) * LDP;
  const __bf16* B1row = BsT + (nc0 + 16 + lm) * LDP;
#pragma unroll
  for (int kc = 0; kc < KC; kc += 32) {
    const v8bf alo = *(const v8bf*)(Arow + kc + hf * 16);
    const v8bf ahi = *(const v8bf*)(Arow + kc + hf * 16 + 8);
    const v16bf a = CAT16(alo, ahi);
    const v8bf b0lo = *(const v8bf*)(B0row + kc + hf * 16);
    const v8bf b0hi = *(const v8bf*)(B0row + kc + hf * 16 + 8);
    const v16bf b0 = CAT16(b0lo, b0hi);
    const v8bf b1lo = *(const v8bf*)(B1row + kc + hf * 16);
    const v8bf b1hi = *(const v8bf*)(B1row + kc + hf * 16 + 8);
    const v16bf b1 = CAT16(b1lo, b1hi);
    acc0 = __builtin_amdgcn_wmma_f32_16x16x32_bf16(false, a, false, b0,
                                                   (short)0, acc0, false, false);
    acc1 = __builtin_amdgcn_wmma_f32_16x16x32_bf16(false, a, false, b1,
                                                   (short)0, acc1, false, false);
  }
}

// ---------------------------------------------------------------------------
// Direct-copy staging: lds[r][kk] = bf16(g[r*gstride + koff + kk]), 64x64 tile.
// Fixed trip count (2 iters/thread), 8 floats/thread/iter:
//   2x global_load_b128 -> 4x v_perm pack -> 1x ds_store_b128.
// ---------------------------------------------------------------------------
__device__ __forceinline__ void stage_f32(const float* __restrict__ g, size_t gstride,
                                          int koff, __bf16* lds, int tid) {
#pragma unroll
  for (int i = 0; i < 2; ++i) {
    const int idx = tid + i * 256;
    const int r = idx >> 3, c8 = (idx & 7) * 8;
    const float* src = g + (size_t)r * gstride + koff + c8;
    const float4 lo = *(const float4*)(src);
    const float4 hi = *(const float4*)(src + 4);
    uint4 p;
    p.x = pkbf(lo.x, lo.y); p.y = pkbf(lo.z, lo.w);
    p.z = pkbf(hi.x, hi.y); p.w = pkbf(hi.z, hi.w);
    *(uint4*)(lds + r * LDP + c8) = p;
  }
}

// ---------------------------------------------------------------------------
// Kernel 1: raw scores S[b,q,k] = sum_d Q[b,q,d]*K[b,k,d]
// grid (LK/64, LQ/64, B), block 256 (8 waves)
// ---------------------------------------------------------------------------
__global__ __launch_bounds__(256)
void k_scores(const float* __restrict__ outp, const float* __restrict__ ctx,
              float* __restrict__ attn) {
  __shared__ __bf16 As[TD * LDP];
  __shared__ __bf16 BsT[TD * LDP];
  const int bi  = blockIdx.z;
  const int qb  = blockIdx.y * TD;
  const int kb  = blockIdx.x * TD;
  const int tid = threadIdx.x;
  const int wv  = tid >> 5;
  const int mr  = (wv >> 1) * 16;
  const int nc0 = (wv & 1) * 32;
  const float* Ag = outp + ((size_t)bi * LQN + qb) * DIMN;
  const float* Bg = ctx  + ((size_t)bi * LKN + kb) * DIMN;
  v8f acc0 = {}, acc1 = {};
  for (int dc = 0; dc < DIMN; dc += KC) {
    stage_f32(Ag, DIMN, dc, As,  tid);
    stage_f32(Bg, DIMN, dc, BsT, tid);
    __syncthreads();
    wmma_tile(As, BsT, mr, nc0, acc0, acc1);
    __syncthreads();
  }
  const int lane = tid & 31, lm = lane & 15, hf = lane >> 4;
  float* Crow = attn + ((size_t)bi * LQN + qb) * LKN + kb;
#pragma unroll
  for (int r = 0; r < 8; ++r) {
    const int m = mr + r + 8 * hf;
    Crow[(size_t)m * LKN + nc0 + lm]      = acc0[r];
    Crow[(size_t)m * LKN + nc0 + 16 + lm] = acc1[r];
  }
}

// ---------------------------------------------------------------------------
// Kernel 2: masked row softmax in-place (one wave per 2048-wide row).
// Rows are 8KB -> stay resident in the 192MB L2 between the three passes.
// ---------------------------------------------------------------------------
__global__ __launch_bounds__(256)
void k_softmax(float* __restrict__ attn, const unsigned char* __restrict__ mask) {
  const int row  = blockIdx.x * 8 + (threadIdx.x >> 5);
  const int lane = threadIdx.x & 31;
  float* s = attn + (size_t)row * LKN;
  const unsigned char* mrow = mask + (size_t)row * LKN;
  float mx = -__builtin_inff();
  for (int k = lane; k < LKN; k += 32)
    if (!mrow[k]) mx = fmaxf(mx, s[k]);
  for (int off = 16; off > 0; off >>= 1) mx = fmaxf(mx, __shfl_xor(mx, off, 32));
  float sum = 0.f;
  for (int k = lane; k < LKN; k += 32)
    if (!mrow[k]) sum += __expf(s[k] - mx);
  for (int off = 16; off > 0; off >>= 1) sum += __shfl_xor(sum, off, 32);
  const float inv = 1.0f / sum;
  for (int k = lane; k < LKN; k += 32)
    s[k] = mrow[k] ? 0.f : __expf(s[k] - mx) * inv;
}

// ---------------------------------------------------------------------------
// Kernel 3: mix[b,q,d] = sum_k P[b,q,k]*K[b,k,d]; stored bf16 in ws.
// B (context) is d-contiguous in memory: staging reads two adjacent k-rows and
// packs (k,k+1) bf16 pairs -> 4B scatter stores into BsT[n][kk].
// grid (D/64, LQ/64, B)
// ---------------------------------------------------------------------------
__global__ __launch_bounds__(256)
void k_mix(const float* __restrict__ attn, const float* __restrict__ ctx,
           __bf16* __restrict__ mixb) {
  __shared__ __bf16 As[TD * LDP];
  __shared__ __bf16 BsT[TD * LDP];
  const int bi  = blockIdx.z;
  const int qb  = blockIdx.y * TD;
  const int db  = blockIdx.x * TD;
  const int tid = threadIdx.x;
  const int wv  = tid >> 5;
  const int mr  = (wv >> 1) * 16;
  const int nc0 = (wv & 1) * 32;
  const float* Ag = attn + ((size_t)bi * LQN + qb) * LKN;
  const float* Bg = ctx  + (size_t)bi * LKN * DIMN + db;
  v8f acc0 = {}, acc1 = {};
  for (int kc0 = 0; kc0 < LKN; kc0 += KC) {
    stage_f32(Ag, LKN, kc0, As, tid);
#pragma unroll
    for (int i = 0; i < 2; ++i) {
      const int idx = tid + i * 256;
      const int kk2 = (idx >> 4) * 2, c4n = (idx & 15) * 4;
      const float* src = Bg + (size_t)(kc0 + kk2) * DIMN + c4n;
      const float4 va = *(const float4*)(src);
      const float4 vb = *(const float4*)(src + DIMN);
      *(uint32_t*)(BsT + (c4n + 0) * LDP + kk2) = pkbf(va.x, vb.x);
      *(uint32_t*)(BsT + (c4n + 1) * LDP + kk2) = pkbf(va.y, vb.y);
      *(uint32_t*)(BsT + (c4n + 2) * LDP + kk2) = pkbf(va.z, vb.z);
      *(uint32_t*)(BsT + (c4n + 3) * LDP + kk2) = pkbf(va.w, vb.w);
    }
    __syncthreads();
    wmma_tile(As, BsT, mr, nc0, acc0, acc1);
    __syncthreads();
  }
  const int lane = tid & 31, lm = lane & 15, hf = lane >> 4;
  __bf16* Crow = mixb + ((size_t)bi * LQN + qb) * DIMN + db;
#pragma unroll
  for (int r = 0; r < 8; ++r) {
    const int m = mr + r + 8 * hf;
    Crow[(size_t)m * DIMN + nc0 + lm]      = f2bf(acc0[r]);
    Crow[(size_t)m * DIMN + nc0 + 16 + lm] = f2bf(acc1[r]);
  }
}

// ---------------------------------------------------------------------------
// Kernel 4: out = tanh([mix, Q] @ W^T + b); K = 2*D, A source switches at D.
// W rows are K-contiguous -> direct staging into BsT.
// grid (D/64, (B*LQ)/64)
// ---------------------------------------------------------------------------
__global__ __launch_bounds__(256)
void k_outproj(const __bf16* __restrict__ mixb, const float* __restrict__ outp,
               const float* __restrict__ W, const float* __restrict__ bias,
               float* __restrict__ outF) {
  __shared__ __bf16 As[TD * LDP];
  __shared__ __bf16 BsT[TD * LDP];
  const int rb  = blockIdx.y * TD;   // flat row in [B*LQ)
  const int nb  = blockIdx.x * TD;   // output feature base
  const int tid = threadIdx.x;
  const int wv  = tid >> 5;
  const int mr  = (wv >> 1) * 16;
  const int nc0 = (wv & 1) * 32;
  v8f acc0 = {}, acc1 = {};
  for (int kc0 = 0; kc0 < 2 * DIMN; kc0 += KC) {
    if (kc0 < DIMN) {
      // A from bf16 mix: contiguous b128 copies
#pragma unroll
      for (int i = 0; i < 2; ++i) {
        const int idx = tid + i * 256;
        const int r = idx >> 3, c8 = (idx & 7) * 8;
        *(v8bf*)(As + r * LDP + c8) =
            *(const v8bf*)(mixb + (size_t)(rb + r) * DIMN + kc0 + c8);
      }
    } else {
      stage_f32(outp + (size_t)rb * DIMN, DIMN, kc0 - DIMN, As, tid);
    }
    stage_f32(W + (size_t)nb * (2 * DIMN), 2 * DIMN, kc0, BsT, tid);
    __syncthreads();
    wmma_tile(As, BsT, mr, nc0, acc0, acc1);
    __syncthreads();
  }
  const int lane = tid & 31, lm = lane & 15, hf = lane >> 4;
#pragma unroll
  for (int r = 0; r < 8; ++r) {
    const int m = mr + r + 8 * hf;
    const float b0 = bias[nb + nc0 + lm];
    const float b1 = bias[nb + nc0 + 16 + lm];
    outF[(size_t)(rb + m) * DIMN + nb + nc0 + lm]      = tanhf(acc0[r] + b0);
    outF[(size_t)(rb + m) * DIMN + nb + nc0 + 16 + lm] = tanhf(acc1[r] + b1);
  }
}

// ---------------------------------------------------------------------------
extern "C" void kernel_launch(void* const* d_in, const int* in_sizes, int n_in,
                              void* d_out, int out_size, void* d_ws, size_t ws_size,
                              hipStream_t stream) {
  const float*         outp = (const float*)d_in[0];          // [B,LQ,D]
  const float*         ctx  = (const float*)d_in[1];          // [B,LK,D]
  const unsigned char* mask = (const unsigned char*)d_in[2];  // [B,LQ,LK] bool
  const float*         W    = (const float*)d_in[3];          // [D,2D]
  const float*         bias = (const float*)d_in[4];          // [D]

  float* outF = (float*)d_out;                                 // out  [B,LQ,D]
  float* attn = (float*)d_out + (size_t)BN * LQN * DIMN;       // attn [B,LQ,LK]
  __bf16* mixb = (__bf16*)d_ws;                                // 16 MB scratch

  dim3 blk(256);
  k_scores <<<dim3(LKN / TD, LQN / TD, BN), blk, 0, stream>>>(outp, ctx, attn);
  k_softmax<<<dim3((BN * LQN) / 8),         blk, 0, stream>>>(attn, mask);
  k_mix    <<<dim3(DIMN / TD, LQN / TD, BN), blk, 0, stream>>>(attn, ctx, mixb);
  k_outproj<<<dim3(DIMN / TD, (BN * LQN) / TD), blk, 0, stream>>>(mixb, outp, W, bias, outF);
}
